// AttentionNetwork_70007966925198
// MI455X (gfx1250) — matrix operand order repaired and checked by reading, not compiled, not run on hardware
//
#include <hip/hip_runtime.h>
#include <hip/hip_bf16.h>
#include <math.h>

typedef __attribute__((ext_vector_type(16))) __bf16 v16bf;
typedef __attribute__((ext_vector_type(8)))  float  v8f;

#define BATCH 16
#define SEQ   1024
#define HID   1024

__device__ __forceinline__ unsigned short f2bf(float f) {
    union { float f; unsigned u; } x; x.f = f;
    unsigned r = x.u + 0x7FFFu + ((x.u >> 16) & 1u);   // round-to-nearest-even
    return (unsigned short)(r >> 16);
}

// term1[b,h] = ht[b,:] . W_attn[h, 0:H] + b_attn[h]
__global__ void k_term1(const float* __restrict__ ht, const float* __restrict__ W_attn,
                        const float* __restrict__ b_attn, float* __restrict__ term1) {
    int t = blockIdx.x * blockDim.x + threadIdx.x;     // BATCH*HID threads
    int b = t >> 10, h = t & 1023;
    const float* hrow = ht + (size_t)b * HID;
    const float* wrow = W_attn + (size_t)h * (2 * HID);
    float acc = b_attn[h];
    for (int k = 0; k < HID; ++k) acc += hrow[k] * wrow[k];
    term1[t] = acc;
}

// Tiled bf16 WMMA GEMM. Block tile 128x64, BK=32, 8 waves (4m x 2n), 2x2 WMMA tiles/wave.
// MODE 0: energy — A = enc [M=B*S, K], B(k,n) = W_attn[n*2H + H + k]; out = tanh(acc + term1[b,n])
// MODE 1: context — A = attn_weights [M=B*S, K], B(k,n) = enc[batch, k, n]; out = acc
template<int MODE>
__global__ void __launch_bounds__(256)
k_gemm(const float* __restrict__ A, const float* __restrict__ Bsrc,
       const float* __restrict__ term1, float* __restrict__ Out) {
    __shared__ unsigned short As[128][34];   // [m][k]
    __shared__ unsigned short Bs[64][34];    // [n][k] (transposed so K-pairs are contiguous)

    const int tid  = threadIdx.x;
    const int lane = tid & 31;
    const int w    = tid >> 5;
    const int wm   = w >> 1;          // 0..3
    const int wn   = w & 1;           // 0..1
    const int lh   = lane >> 4;       // lane half
    const int ll   = lane & 15;
    const int tileM = blockIdx.x * 128;
    const int tileN = blockIdx.y * 64;
    const int batch = tileM >> 10;    // M tiles never cross a batch (128 | 1024)

    const v8f vzero = {0.f,0.f,0.f,0.f,0.f,0.f,0.f,0.f};
    v8f acc[2][2];
    #pragma unroll
    for (int i = 0; i < 2; ++i)
        #pragma unroll
        for (int j = 0; j < 2; ++j) acc[i][j] = vzero;

    union Frag { unsigned u[8]; v16bf v; };

    for (int k0 = 0; k0 < HID; k0 += 32) {
        // stage A tile 128x32 (fp32 -> bf16), coalesced along k
        for (int idx = tid; idx < 128 * 32; idx += 256) {
            int r = idx >> 5, c = idx & 31;
            As[r][c] = f2bf(A[(size_t)(tileM + r) * HID + (k0 + c)]);
        }
        // stage B tile 32x64 -> Bs[n][k]
        for (int idx = tid; idx < 32 * 64; idx += 256) {
            int kk = idx >> 6, n = idx & 63;
            float v;
            if (MODE == 0)
                v = Bsrc[(size_t)(tileN + n) * (2 * HID) + HID + (k0 + kk)];   // W2 (L2-resident)
            else
                v = Bsrc[(size_t)batch * (SEQ * HID) + (size_t)(k0 + kk) * HID + (tileN + n)];
            Bs[n][kk] = f2bf(v);
        }
        __syncthreads();

        Frag a[2], bb[2];
        #pragma unroll
        for (int i = 0; i < 2; ++i) {   // A 16x32 bf16 layout: lane=M, VGPR0-3: K=lh*8+2j, VGPR4-7: +16
            const unsigned* arow = (const unsigned*)&As[wm * 32 + i * 16 + ll][0];
            #pragma unroll
            for (int j = 0; j < 4; ++j) {
                a[i].u[j]     = arow[lh * 4 + j];
                a[i].u[4 + j] = arow[8 + lh * 4 + j];
            }
        }
        #pragma unroll
        for (int j = 0; j < 2; ++j) {   // B 32x16 bf16 layout: lane=N, VGPR0-3: K=lh*8+2v, VGPR4-7: +16
            const unsigned* brow = (const unsigned*)&Bs[wn * 32 + j * 16 + ll][0];
            #pragma unroll
            for (int v = 0; v < 8; ++v) bb[j].u[v] = brow[(v >> 2) * 8 + lh * 4 + (v & 3)];
        }

        #pragma unroll
        for (int i = 0; i < 2; ++i)
            #pragma unroll
            for (int j = 0; j < 2; ++j)
                acc[i][j] = __builtin_amdgcn_wmma_f32_16x16x32_bf16(
                    false, a[i].v, false, bb[j].v, (short)0, acc[i][j], false, false);
        __syncthreads();
    }

    // epilogue; C/D layout: VGPR r -> row r + 8*lh, col = ll
    #pragma unroll
    for (int i = 0; i < 2; ++i) {
        #pragma unroll
        for (int j = 0; j < 2; ++j) {
            int n = tileN + wn * 32 + j * 16 + ll;
            #pragma unroll
            for (int r = 0; r < 8; ++r) {
                int m = tileM + wm * 32 + i * 16 + r + 8 * lh;
                float v = acc[i][j][r];
                if (MODE == 0) v = tanhf(v + term1[batch * HID + n]);
                Out[(size_t)m * HID + n] = v;
            }
        }
    }
}

// column softmax stats over the seq axis: per (b,h) max and 1/sum(exp)
__global__ void k_colstats(const float* __restrict__ E,
                           float* __restrict__ mstat, float* __restrict__ rstat) {
    int t = blockIdx.x * blockDim.x + threadIdx.x;     // BATCH*HID threads
    int b = t >> 10, h = t & 1023;
    const float* col = E + (size_t)b * (SEQ * HID) + h;
    float m = -3.402823466e38f;
    for (int s = 0; s < SEQ; ++s) m = fmaxf(m, col[(size_t)s * HID]);
    float sum = 0.f;
    for (int s = 0; s < SEQ; ++s) sum += __expf(col[(size_t)s * HID] - m);
    mstat[t] = m;
    rstat[t] = 1.0f / sum;
}

__global__ void k_softmax_norm(float* __restrict__ E,
                               const float* __restrict__ mstat,
                               const float* __restrict__ rstat) {
    size_t idx = (size_t)blockIdx.x * blockDim.x + threadIdx.x;
    int h  = (int)(idx & 1023);
    int b  = (int)(idx >> 20);                          // SEQ*HID = 1<<20 per batch
    int bh = b * HID + h;
    E[idx] = __expf(E[idx] - mstat[bh]) * rstat[bh];
}

extern "C" void kernel_launch(void* const* d_in, const int* in_sizes, int n_in,
                              void* d_out, int out_size, void* d_ws, size_t ws_size,
                              hipStream_t stream) {
    const float* ht     = (const float*)d_in[0];
    const float* enc    = (const float*)d_in[1];
    const float* W_attn = (const float*)d_in[2];
    const float* b_attn = (const float*)d_in[3];
    // d_in[4] = W_v : dead output in the reference, unused

    const size_t NT = (size_t)BATCH * SEQ * HID;       // 16M elements
    float* context = (float*)d_out;                    // output 0: [B,S,H]
    float* attnw   = (float*)d_out + NT;               // output 1: energy -> softmax in place

    float* term1 = (float*)d_ws;                       // [B*H]
    float* mstat = term1 + BATCH * HID;                // [B*H]
    float* rstat = mstat + BATCH * HID;                // [B*H]

    k_term1<<<(BATCH * HID) / 256, 256, 0, stream>>>(ht, W_attn, b_attn, term1);

    dim3 g1((BATCH * SEQ) / 128, HID / 64);
    k_gemm<0><<<g1, 256, 0, stream>>>(enc, W_attn, term1, attnw);

    k_colstats<<<(BATCH * HID) / 256, 256, 0, stream>>>(attnw, mstat, rstat);
    k_softmax_norm<<<(unsigned)(NT / 256), 256, 0, stream>>>(attnw, mstat, rstat);

    k_gemm<1><<<g1, 256, 0, stream>>>(attnw, enc, term1, context);
}
